// BERT_43284680409112
// MI455X (gfx1250) — compile-verified, matrix operand order checked by reading, not run on hardware
//
#include <hip/hip_runtime.h>
#include <hip/hip_bf16.h>
#include <math.h>

// ---------------------------------------------------------------------------
// BERT single-layer forward on gfx1250 (CDNA5, wave32, WMMA + async LDS).
//
// All GEMM operands live in f16 workspace buffers; every B operand is stored
// N x K (transposed at conversion time) so tile staging is a pure contiguous
// copy. The GEMM stages tiles with GLOBAL_LOAD_ASYNC_TO_LDS_B128 (ASYNCcnt),
// ping-pong double buffered: issue async copies for tile i+1, run tile i's
// WMMAs, s_wait_asynccnt 0 + barrier. Fragments are read as 2x ds_load_b128
// (ISA-contiguous per-lane layout); accumulation is f32 via
// v_wmma_f32_16x16x32_f16; epilogue fuses alpha / f32-bias / exact-GELU.
//
// CONTRACT (all call sites satisfy it): M % 32 == 0, N % 64 == 0,
// K % 64 == 0, lda/ldb/ldc % 8 == 0, batch strides % 8 == 0.
// ---------------------------------------------------------------------------

#define BB 4
#define SS 512
#define DD 768
#define EE 768
#define HH 12
#define VV 32000
#define LN_EPS 1e-5f

typedef __attribute__((ext_vector_type(4)))  _Float16 v4h;
typedef __attribute__((ext_vector_type(8)))  _Float16 v8h;
typedef __attribute__((ext_vector_type(16))) _Float16 v16h;
typedef __attribute__((ext_vector_type(8)))  float    v8f;

// Issue one per-lane 16-byte global -> LDS async copy (tracked by ASYNCcnt).
__device__ __forceinline__ void async_copy_b128(const _Float16* gptr, void* lptr)
{
    unsigned lds = (unsigned)(unsigned long long)lptr;          // LDS offset
    unsigned long long ga = (unsigned long long)gptr;           // 64-bit VA
    asm volatile("global_load_async_to_lds_b128 %0, %1, off"
                 :: "v"(lds), "v"(ga) : "memory");
}
__device__ __forceinline__ void wait_async_zero()
{
    asm volatile("s_wait_asynccnt 0x0" ::: "memory");
}

// ---------------------------------------------------------------------------
// Transpose-convert: out[C][R] (f16) = in[R][C] (TIn), LDS-tiled, batched.
// Requires R % 32 == 0 and C % 32 == 0.
// ---------------------------------------------------------------------------
template <typename TIn>
__global__ __launch_bounds__(256)
void transpose_cvt_kernel(const TIn* __restrict__ in, _Float16* __restrict__ out,
                          int R, int C, long long strideIn, long long strideOut)
{
    __shared__ _Float16 tile[32][34];
    const int z = blockIdx.z;
    in  += (long long)z * strideIn;
    out += (long long)z * strideOut;
    const int r0 = blockIdx.y * 32, c0 = blockIdx.x * 32;
    const int tr = threadIdx.x >> 5;   // 0..7
    const int tc = threadIdx.x & 31;
    #pragma unroll
    for (int p = 0; p < 4; ++p) {
        const int r = r0 + tr + p * 8;
        tile[tr + p * 8][tc] = (_Float16)in[(long long)r * C + (c0 + tc)];
    }
    __syncthreads();
    #pragma unroll
    for (int p = 0; p < 4; ++p) {
        const int c = c0 + tr + p * 8;
        out[(long long)c * R + (r0 + tc)] = tile[tc][tr + p * 8];
    }
}

// ---------------------------------------------------------------------------
// Batched WMMA GEMM, f16 operands / f32 accumulate.
//   C[z] = act( alpha * A[z] @ B[z]^T + bias[z] ),  B stored N x K.
// Block tile 32(M) x 64(N), BK = 64, 256 threads = 8 waves, one 16x16
// accumulator per wave, async-copy double-buffered LDS.
// ---------------------------------------------------------------------------
__global__ __launch_bounds__(256)
void gemm_wmma_f16(const _Float16* __restrict__ A, int lda, long long strideA,
                   const _Float16* __restrict__ Bm, int ldb, long long strideB,
                   void* __restrict__ Cv, int ldc, long long strideC, int outF16,
                   const float* __restrict__ bias, long long strideBias,
                   int hasBias, int K, float alpha, int act)
{
    // Row pitch 72 halves = 144 B (multiple of 16 B -> aligned b128 access).
    __shared__ __align__(16) _Float16 As[2][32][72];
    __shared__ __align__(16) _Float16 Bt[2][64][72];   // B tile, [n][k]

    const int z = blockIdx.z;
    A  += (long long)z * strideA;
    Bm += (long long)z * strideB;

    const int m0 = blockIdx.y * 32;
    const int n0 = blockIdx.x * 64;

    const int tid   = threadIdx.x;
    const int lane  = tid & 31;
    const int wave  = tid >> 5;
    const int waveM = wave >> 2;               // 0..1
    const int waveN = wave & 3;                // 0..3

    const int mr  = 16 * waveM + (lane & 15);  // A fragment row in LDS
    const int nc  = 16 * waveN + (lane & 15);  // B fragment row in Bt
    const int khi = (lane >= 16) ? 8 : 0;      // per-lane K half-offset

    // Per-thread staging coordinates (one b128 = 8 halves per copy).
    const int ar = tid >> 3;                   // A row 0..31
    const int ac = (tid & 7) << 3;             // A col (k), step 8
    const int br0 = tid >> 3;                  // B rows br0 and br0+32
    const int bk0 = (tid & 7) << 3;

    // Issue the 3 async b128 copies that stage one 32x64 A + 64x64 B tile.
    auto stage = [&](int sb, int kks) {
        async_copy_b128(A + (long long)(m0 + ar) * lda + (kks + ac),
                        &As[sb][ar][ac]);
        async_copy_b128(Bm + (long long)(n0 + br0) * ldb + (kks + bk0),
                        &Bt[sb][br0][bk0]);
        async_copy_b128(Bm + (long long)(n0 + br0 + 32) * ldb + (kks + bk0),
                        &Bt[sb][br0 + 32][bk0]);
    };

    v8f acc = {};

    stage(0, 0);
    wait_async_zero();
    __syncthreads();

    int sb = 0;
    for (int kk = 0; kk < K; kk += 64) {
        // Prefetch the tile after next (branchless; gfx1250 global_prefetch).
        const int kpf = (kk + 128 < K) ? (kk + 128) : kk;
        __builtin_prefetch(A + (long long)(m0 + ar) * lda + (kpf + ac), 0, 1);

        // Kick off next tile's async copies; no wait -> overlap with WMMAs.
        if (kk + 64 < K) stage(sb ^ 1, kk + 64);

        // Two 16x16x32 WMMAs over k halves [0,32) and [32,64) of buffer sb.
        // ISA fragment layout (wave32, 16-bit): lane L holds row L&15, halves
        // 0..7 = K[khi..khi+8), halves 8..15 = K[khi+16..khi+24) ->
        // two contiguous 16-byte LDS reads per fragment.
        #pragma unroll
        for (int j = 0; j < 2; ++j) {
            const int kb = 32 * j + khi;
            const v8h alo = *(const v8h*)&As[sb][mr][kb];
            const v8h ahi = *(const v8h*)&As[sb][mr][kb + 16];
            const v8h blo = *(const v8h*)&Bt[sb][nc][kb];
            const v8h bhi = *(const v8h*)&Bt[sb][nc][kb + 16];
            const v16h afrag = __builtin_shufflevector(alo, ahi,
                0, 1, 2, 3, 4, 5, 6, 7, 8, 9, 10, 11, 12, 13, 14, 15);
            const v16h bfrag = __builtin_shufflevector(blo, bhi,
                0, 1, 2, 3, 4, 5, 6, 7, 8, 9, 10, 11, 12, 13, 14, 15);
            acc = __builtin_amdgcn_wmma_f32_16x16x32_f16(
                false, afrag, false, bfrag, (short)0, acc, false, false);
        }

        wait_async_zero();   // own async copies done (no-op on last iter)
        __syncthreads();     // all waves' copies visible
        sb ^= 1;
    }

    // Epilogue. C/D layout: lane -> N = lane&15, VGPR v -> M = v + khi.
    const int gn  = n0 + nc;
    const int gm0 = m0 + 16 * waveM + khi;
    const float bval = hasBias ? bias[(long long)z * strideBias + gn] : 0.f;
    if (outF16) {
        _Float16* C = (_Float16*)Cv + (long long)z * strideC;
        #pragma unroll
        for (int v = 0; v < 8; ++v) {
            float val = acc[v] * alpha + bval;
            if (act == 1)
                val = 0.5f * val * (1.0f + erff(val * 0.70710678118654752f));
            C[(long long)(gm0 + v) * ldc + gn] = (_Float16)val;
        }
    } else {
        float* C = (float*)Cv + (long long)z * strideC;
        #pragma unroll
        for (int v = 0; v < 8; ++v) {
            float val = acc[v] * alpha + bval;
            if (act == 1)
                val = 0.5f * val * (1.0f + erff(val * 0.70710678118654752f));
            C[(long long)(gm0 + v) * ldc + gn] = val;
        }
    }
}

// ---------------------------------------------------------------------------
// Embedding (token + segment + sinusoidal positional) fused with LayerNorm.
// Emits f16 (consumed by the QKV GEMMs). One block per token row.
// ---------------------------------------------------------------------------
__global__ __launch_bounds__(256)
void embed_ln_kernel(const int* __restrict__ tok,
                     const float* __restrict__ tok_emb,
                     const float* __restrict__ seg_emb,
                     const float* __restrict__ g,
                     const float* __restrict__ b,
                     _Float16* __restrict__ x16)
{
    __shared__ float row[DD];
    __shared__ float red[256];
    const int r = blockIdx.x;            // 0 .. B*S-1
    const int s = r % SS;
    const int id  = tok[r];
    const int seg = (s >= SS / 2 + 1) ? 1 : 0;
    const int tid = threadIdx.x;

    float sum = 0.f;
    for (int d = tid; d < DD; d += 256) {
        float ang = (float)s / powf(10000.0f, 2.0f * (float)d / (float)DD);
        float pos = ((d & 1) == 0) ? sinf(ang) : cosf(ang);
        float v = tok_emb[(long long)id * DD + d] + seg_emb[(long long)seg * DD + d] + pos;
        row[d] = v;
        sum += v;
    }
    red[tid] = sum; __syncthreads();
    for (int o = 128; o > 0; o >>= 1) {
        if (tid < o) red[tid] += red[tid + o];
        __syncthreads();
    }
    const float mean = red[0] / (float)DD;
    __syncthreads();

    float vs = 0.f;
    for (int d = tid; d < DD; d += 256) { float t = row[d] - mean; vs += t * t; }
    red[tid] = vs; __syncthreads();
    for (int o = 128; o > 0; o >>= 1) {
        if (tid < o) red[tid] += red[tid + o];
        __syncthreads();
    }
    const float rstd = rsqrtf(red[0] / (float)DD + LN_EPS);
    __syncthreads();

    for (int d = tid; d < DD; d += 256)
        x16[(long long)r * DD + d] = (_Float16)((row[d] - mean) * rstd * g[d] + b[d]);
}

// ---------------------------------------------------------------------------
// LayerNorm over last dim (width DD): f32 in, f16 out. One block per row.
// ---------------------------------------------------------------------------
__global__ __launch_bounds__(256)
void layernorm_f16_kernel(const float* __restrict__ xin,
                          _Float16* __restrict__ xout,
                          const float* __restrict__ g,
                          const float* __restrict__ b)
{
    __shared__ float row[DD];
    __shared__ float red[256];
    const int r = blockIdx.x;
    const int tid = threadIdx.x;
    const float* xr = xin + (long long)r * DD;

    float sum = 0.f;
    for (int d = tid; d < DD; d += 256) { float v = xr[d]; row[d] = v; sum += v; }
    red[tid] = sum; __syncthreads();
    for (int o = 128; o > 0; o >>= 1) {
        if (tid < o) red[tid] += red[tid + o];
        __syncthreads();
    }
    const float mean = red[0] / (float)DD;
    __syncthreads();

    float vs = 0.f;
    for (int d = tid; d < DD; d += 256) { float t = row[d] - mean; vs += t * t; }
    red[tid] = vs; __syncthreads();
    for (int o = 128; o > 0; o >>= 1) {
        if (tid < o) red[tid] += red[tid + o];
        __syncthreads();
    }
    const float rstd = rsqrtf(red[0] / (float)DD + LN_EPS);
    __syncthreads();

    for (int d = tid; d < DD; d += 256)
        xout[(long long)r * DD + d] = (_Float16)((row[d] - mean) * rstd * g[d] + b[d]);
}

// ---------------------------------------------------------------------------
// Masked softmax over f32 score rows of length SS, emits f16 attention
// (consumed by the ctx GEMM). blockIdx.x = (h*B + b)*S + s.
// ---------------------------------------------------------------------------
__global__ __launch_bounds__(256)
void softmax_mask_kernel(const float* __restrict__ scores,
                         _Float16* __restrict__ attn,
                         const unsigned char* __restrict__ mask)
{
    __shared__ float buf[SS];
    __shared__ float red[256];
    const long long rid = blockIdx.x;
    const int s = (int)(rid % SS);
    const int b = (int)((rid / SS) % BB);
    const float* row = scores + rid * SS;
    _Float16* orow = attn + rid * SS;
    const unsigned char* mrow = mask + ((long long)b * SS + s) * SS;
    const int tid = threadIdx.x;

    float mx = -3.4e38f;
    for (int t = tid; t < SS; t += 256) {
        float v = mrow[t] ? -1e9f : row[t];
        buf[t] = v;
        mx = fmaxf(mx, v);
    }
    red[tid] = mx; __syncthreads();
    for (int o = 128; o > 0; o >>= 1) {
        if (tid < o) red[tid] = fmaxf(red[tid], red[tid + o]);
        __syncthreads();
    }
    const float rowmax = red[0];
    __syncthreads();

    float sum = 0.f;
    for (int t = tid; t < SS; t += 256) {
        float e = expf(buf[t] - rowmax);
        buf[t] = e;
        sum += e;
    }
    red[tid] = sum; __syncthreads();
    for (int o = 128; o > 0; o >>= 1) {
        if (tid < o) red[tid] += red[tid + o];
        __syncthreads();
    }
    const float inv = 1.0f / red[0];
    __syncthreads();

    for (int t = tid; t < SS; t += 256)
        orow[t] = (_Float16)(buf[t] * inv);
}

// ---------------------------------------------------------------------------
// Concatenate heads: ctx16 [H][B][S][E] -> ctx2 [B*S][H*E] (f16, coalesced
// writes).
// ---------------------------------------------------------------------------
__global__ __launch_bounds__(256)
void concat_heads_kernel(const _Float16* __restrict__ ctx,
                         _Float16* __restrict__ ctx2)
{
    const long long total = (long long)BB * SS * HH * EE;
    long long i = (long long)blockIdx.x * 256 + threadIdx.x;
    if (i >= total) return;
    int e = (int)(i % EE);
    long long t = i / EE;
    int h = (int)(t % HH);
    t /= HH;
    int s = (int)(t % SS);
    int b = (int)(t / SS);
    ctx2[i] = ctx[((((long long)h * BB + b) * SS + s) * EE) + e];
}

// ---------------------------------------------------------------------------
// In-place log-softmax over rows of length VV. One block per row.
// ---------------------------------------------------------------------------
__global__ __launch_bounds__(256)
void log_softmax_kernel(float* __restrict__ x)
{
    __shared__ float red[256];
    const long long r = blockIdx.x;
    float* row = x + r * (long long)VV;
    const int tid = threadIdx.x;

    float mx = -3.4e38f;
    for (int i = tid; i < VV; i += 256) mx = fmaxf(mx, row[i]);
    red[tid] = mx; __syncthreads();
    for (int o = 128; o > 0; o >>= 1) {
        if (tid < o) red[tid] = fmaxf(red[tid], red[tid + o]);
        __syncthreads();
    }
    const float rowmax = red[0];
    __syncthreads();

    float sum = 0.f;
    for (int i = tid; i < VV; i += 256) sum += expf(row[i] - rowmax);
    red[tid] = sum; __syncthreads();
    for (int o = 128; o > 0; o >>= 1) {
        if (tid < o) red[tid] += red[tid + o];
        __syncthreads();
    }
    const float lse = logf(red[0]) + rowmax;
    __syncthreads();

    for (int i = tid; i < VV; i += 256) row[i] = row[i] - lse;
}

// ---------------------------------------------------------------------------
// Classifier head: cls[b][j] = enc16[b, s=0, :] @ Wc[:, j] + bc[j].
// ---------------------------------------------------------------------------
__global__ void cls_kernel(const _Float16* __restrict__ enc,
                           const float* __restrict__ Wc,
                           const float* __restrict__ bc,
                           float* __restrict__ out)
{
    const int t = threadIdx.x;
    if (t < BB * 2) {
        const int b = t >> 1, j = t & 1;
        float s = bc[j];
        const _Float16* er = enc + (long long)b * SS * DD;   // row s = 0
        for (int d = 0; d < DD; ++d) s += (float)er[d] * Wc[d * 2 + j];
        out[t] = s;
    }
}

// ---------------------------------------------------------------------------
// Host-side orchestration.
// ---------------------------------------------------------------------------
static inline void launch_gemm(hipStream_t stream,
                               const _Float16* A, int lda, long long strideA,
                               const _Float16* Bm, int ldb, long long strideB,
                               void* C, int ldc, long long strideC, int outF16,
                               const float* bias, long long strideBias, int hasBias,
                               int M, int N, int K, float alpha, int act, int batch)
{
    dim3 grid((N + 63) / 64, (M + 31) / 32, batch);
    gemm_wmma_f16<<<grid, 256, 0, stream>>>(A, lda, strideA, Bm, ldb, strideB,
                                            C, ldc, strideC, outF16,
                                            bias, strideBias, hasBias,
                                            K, alpha, act);
}

// Transpose-convert launcher: out[C][R] f16 = in[R][C].
template <typename TIn>
static inline void launch_tcvt(hipStream_t stream, const TIn* in, _Float16* out,
                               int R, int C, long long strideIn,
                               long long strideOut, int batch)
{
    dim3 grid(C / 32, R / 32, batch);
    transpose_cvt_kernel<TIn><<<grid, 256, 0, stream>>>(in, out, R, C,
                                                        strideIn, strideOut);
}

extern "C" void kernel_launch(void* const* d_in, const int* in_sizes, int n_in,
                              void* d_out, int out_size, void* d_ws, size_t ws_size,
                              hipStream_t stream)
{
    (void)in_sizes; (void)n_in; (void)out_size; (void)ws_size;

    const int*           input_tensor = (const int*)d_in[0];
    const unsigned char* attn_mask    = (const unsigned char*)d_in[1];
    const float* tok_emb  = (const float*)d_in[2];
    const float* seg_emb  = (const float*)d_in[3];
    const float* ln_emb_g = (const float*)d_in[4];
    const float* ln_emb_b = (const float*)d_in[5];
    const float* Wq = (const float*)d_in[6];
    const float* bq = (const float*)d_in[7];
    const float* Wk = (const float*)d_in[8];
    const float* bk = (const float*)d_in[9];
    const float* Wv = (const float*)d_in[10];
    const float* bv = (const float*)d_in[11];
    const float* Wo = (const float*)d_in[12];
    const float* bo = (const float*)d_in[13];
    const float* ln_attn_g = (const float*)d_in[14];
    const float* ln_attn_b = (const float*)d_in[15];
    const float* W1 = (const float*)d_in[16];
    const float* b1 = (const float*)d_in[17];
    const float* W2 = (const float*)d_in[18];
    const float* b2 = (const float*)d_in[19];
    const float* ln_ff_g = (const float*)d_in[20];
    const float* ln_ff_b = (const float*)d_in[21];
    const float* Wp = (const float*)d_in[22];
    const float* bp = (const float*)d_in[23];
    const float* Wc = (const float*)d_in[24];
    const float* bc = (const float*)d_in[25];

    float* out = (float*)d_out;

    // element counts
    const long long XSZ = (long long)BB * SS * DD;        // 1,572,864
    const long long QSZ = (long long)HH * BB * SS * EE;   // 18,874,368
    const long long SSZ = (long long)HH * BB * SS * SS;   // 12,582,912
    const long long WQK = (long long)HH * DD * EE;        // 7,077,888
    const long long WPN = (long long)DD * VV;             // 24,576,000

    // carve workspace (256 B aligned slabs)
    char* base = (char*)d_ws;
    auto carve = [&](long long bytes) -> char* {
        char* p = base;
        base += (bytes + 255) & ~(long long)255;
        return p;
    };
    _Float16* wq16T  = (_Float16*)carve(WQK * 2);        // [H][E][D]
    _Float16* wk16T  = (_Float16*)carve(WQK * 2);
    _Float16* wv16T  = (_Float16*)carve(WQK * 2);
    _Float16* wo16T  = (_Float16*)carve(WQK * 2);        // [D][H*E]
    _Float16* w1T    = (_Float16*)carve((long long)DD * EE * 2);  // [E][D]
    _Float16* w2T    = (_Float16*)carve((long long)EE * DD * 2);  // [D][E]
    _Float16* wpT    = (_Float16*)carve(WPN * 2);        // [V][D]
    _Float16* x16    = (_Float16*)carve(XSZ * 2);
    _Float16* q16    = (_Float16*)carve(QSZ * 2);        // [H][B*S][E]
    _Float16* k16    = (_Float16*)carve(QSZ * 2);        // [H][B*S][E]
    _Float16* v16    = (_Float16*)carve(QSZ * 2);        // [H][B*S][E]
    _Float16* vT     = (_Float16*)carve(QSZ * 2);        // [H*B][E][S]
    float*    sc     = (float*)   carve(SSZ * 4);
    _Float16* attn16 = (_Float16*)carve(SSZ * 2);
    _Float16* ctx16  = (_Float16*)carve(QSZ * 2);
    _Float16* ctx2   = (_Float16*)carve(QSZ * 2);
    float*    ao32   = (float*)   carve(XSZ * 4);
    _Float16* aoln   = (_Float16*)carve(XSZ * 2);
    _Float16* h1_16  = (_Float16*)carve((long long)BB * SS * EE * 2);
    float*    enc32  = (float*)   carve(XSZ * 4);
    _Float16* encln  = (_Float16*)carve(XSZ * 2);

    const int ROWS = BB * SS;  // 2048

    // 0. convert + transpose weights to f16 N x K layout (once per launch)
    launch_tcvt<float>(stream, Wq, wq16T, DD, EE, (long long)DD * EE,
                       (long long)DD * EE, HH);
    launch_tcvt<float>(stream, Wk, wk16T, DD, EE, (long long)DD * EE,
                       (long long)DD * EE, HH);
    launch_tcvt<float>(stream, Wv, wv16T, DD, EE, (long long)DD * EE,
                       (long long)DD * EE, HH);
    launch_tcvt<float>(stream, Wo, wo16T, HH * EE, DD, 0, 0, 1);
    launch_tcvt<float>(stream, W1, w1T, DD, EE, 0, 0, 1);
    launch_tcvt<float>(stream, W2, w2T, EE, DD, 0, 0, 1);
    launch_tcvt<float>(stream, Wp, wpT, DD, VV, 0, 0, 1);

    // 1. embedding + LN -> f16 x
    embed_ln_kernel<<<ROWS, 256, 0, stream>>>(input_tensor, tok_emb, seg_emb,
                                              ln_emb_g, ln_emb_b, x16);

    // 2. Q/K/V projections, batched over heads: [H][B*S][E], f16 out
    launch_gemm(stream, x16, DD, 0, wq16T, DD, (long long)DD * EE,
                q16, EE, (long long)ROWS * EE, 1, bq, EE, 1,
                ROWS, EE, DD, 1.0f, 0, HH);
    launch_gemm(stream, x16, DD, 0, wk16T, DD, (long long)DD * EE,
                k16, EE, (long long)ROWS * EE, 1, bk, EE, 1,
                ROWS, EE, DD, 1.0f, 0, HH);
    launch_gemm(stream, x16, DD, 0, wv16T, DD, (long long)DD * EE,
                v16, EE, (long long)ROWS * EE, 1, bv, EE, 1,
                ROWS, EE, DD, 1.0f, 0, HH);

    // 3. scores = Q @ K^T / sqrt(S): K is already N x K; 48 batches, f32 out
    const float inv_sqrt_s = 1.0f / sqrtf((float)SS);
    launch_gemm(stream, q16, EE, (long long)SS * EE,
                k16, EE, (long long)SS * EE,
                sc, SS, (long long)SS * SS, 0, nullptr, 0, 0,
                SS, SS, EE, inv_sqrt_s, 0, HH * BB);

    // 4. mask + softmax -> f16 attention
    softmax_mask_kernel<<<HH * BB * SS, 256, 0, stream>>>(sc, attn16, attn_mask);

    // 4b. transpose V per (h,b): [S][E] -> [E][S] so ctx's B is N x K
    launch_tcvt<_Float16>(stream, v16, vT, SS, EE, (long long)SS * EE,
                          (long long)SS * EE, HH * BB);

    // 5. ctx = attn @ V: B = vT [E][S]; 48 batches, f16 out
    launch_gemm(stream, attn16, SS, (long long)SS * SS,
                vT, SS, (long long)EE * SS,
                ctx16, EE, (long long)SS * EE, 1, nullptr, 0, 0,
                SS, EE, SS, 1.0f, 0, HH * BB);

    // 6. concat heads -> [B*S][H*E] (f16)
    {
        const long long total = (long long)BB * SS * HH * EE;
        concat_heads_kernel<<<(unsigned)((total + 255) / 256), 256, 0, stream>>>(ctx16, ctx2);
    }

    // 7. output projection + bias (f32 out), then LN -> f16
    launch_gemm(stream, ctx2, HH * EE, 0, wo16T, HH * EE, 0,
                ao32, DD, 0, 0, bo, 0, 1,
                ROWS, DD, HH * EE, 1.0f, 0, 1);
    layernorm_f16_kernel<<<ROWS, 256, 0, stream>>>(ao32, aoln, ln_attn_g, ln_attn_b);

    // 8. FFN: gelu(aoln @ W1 + b1) -> f16; @ W2 + b2 -> f32; LN -> f16
    launch_gemm(stream, aoln, DD, 0, w1T, DD, 0,
                h1_16, EE, 0, 1, b1, 0, 1,
                ROWS, EE, DD, 1.0f, 1, 1);
    launch_gemm(stream, h1_16, EE, 0, w2T, EE, 0,
                enc32, DD, 0, 0, b2, 0, 1,
                ROWS, DD, EE, 1.0f, 0, 1);
    layernorm_f16_kernel<<<ROWS, 256, 0, stream>>>(enc32, encln, ln_ff_g, ln_ff_b);

    // 9. vocab projection straight into d_out (f32), then in-place log-softmax
    launch_gemm(stream, encln, DD, 0, wpT, DD, 0,
                out, VV, 0, 0, bp, 0, 1,
                ROWS, VV, DD, 1.0f, 0, 1);
    log_softmax_kernel<<<ROWS, 256, 0, stream>>>(out);

    // 10. classifier head -> last 8 floats of d_out
    cls_kernel<<<1, 32, 0, stream>>>(encln, Wc, bc, out + (long long)ROWS * VV);
}
